// ARDRBFKernelOld_33483565040049
// MI455X (gfx1250) — compile-verified, matrix operand order checked by reading, not run on hardware
//
#include <hip/hip_runtime.h>

// ---------------------------------------------------------------------------
// ARD-RBF kernel for MI455X (gfx1250, wave32, WMMA f16 16x16x32)
//   out[i,j] = var * exp(-0.5 * (||a_i||^2 - 2 a_i.b_j + ||b_j||^2))
//   a = x1 * sqrt(softmax(alpha_raw^2)),  b = x2 * sqrt(...)
// N = M = 8192, D = 256
// ---------------------------------------------------------------------------

typedef _Float16 v8h  __attribute__((ext_vector_type(8)));
typedef _Float16 v16h __attribute__((ext_vector_type(16)));
typedef float    v8f  __attribute__((ext_vector_type(8)));
typedef float    v4f  __attribute__((ext_vector_type(4)));

#define D_DIM  256
#define N_ROWS 8192
#define M_ROWS 8192

// ---------------------------------------------------------------------------
// Kernel 1: alpha = softmax(alpha_raw^2); sa = sqrt(alpha); var = variance^2
// Single block of 256 threads (D = 256).
// ---------------------------------------------------------------------------
__global__ __launch_bounds__(256) void prep_alpha(const float* __restrict__ alpha_raw,
                                                  const float* __restrict__ variance_raw,
                                                  float* __restrict__ sa,
                                                  float* __restrict__ var_out) {
    __shared__ float red[256];
    const int t = threadIdx.x;
    const float s = alpha_raw[t] * alpha_raw[t];

    red[t] = s;
    __syncthreads();
    for (int off = 128; off > 0; off >>= 1) {
        if (t < off) red[t] = fmaxf(red[t], red[t + off]);
        __syncthreads();
    }
    const float mx = red[0];
    __syncthreads();

    const float e = __expf(s - mx);
    red[t] = e;
    __syncthreads();
    for (int off = 128; off > 0; off >>= 1) {
        if (t < off) red[t] += red[t + off];
        __syncthreads();
    }
    const float sum = red[0];

    sa[t] = sqrtf(e / sum);
    if (t == 0) var_out[0] = variance_raw[0] * variance_raw[0];
}

// ---------------------------------------------------------------------------
// Kernel 2: scale rows by sqrt(alpha), convert to f16 panels, and compute
// fp32 row norms FROM THE ROUNDED f16 VALUES (keeps sq consistent & >= 0).
// One wave32 per row; 8 rows per 256-thread block. 16384 rows total.
// ---------------------------------------------------------------------------
__global__ __launch_bounds__(256) void scale_convert(const float* __restrict__ x1,
                                                     const float* __restrict__ x2,
                                                     const float* __restrict__ sa,
                                                     _Float16* __restrict__ a_h,
                                                     _Float16* __restrict__ b_h,
                                                     float* __restrict__ n1,
                                                     float* __restrict__ n2) {
    const int wave = threadIdx.x >> 5;
    const int lane = threadIdx.x & 31;
    const long row = (long)blockIdx.x * 8 + wave;   // 0 .. 16383

    const float* src;
    _Float16*    dst;
    float*       nrm;
    long         r;
    if (row < N_ROWS) { src = x1; dst = a_h; nrm = n1; r = row; }
    else              { src = x2; dst = b_h; nrm = n2; r = row - N_ROWS; }

    const float* p  = src + r * D_DIM + lane * 8;
    const float* sp = sa + lane * 8;
    const v4f xa = *(const v4f*)(p);
    const v4f xb = *(const v4f*)(p + 4);
    const v4f s0 = *(const v4f*)(sp);
    const v4f s1 = *(const v4f*)(sp + 4);

    float acc = 0.0f;
    v8h h;
#pragma unroll
    for (int i = 0; i < 4; ++i) {
        const _Float16 hv = (_Float16)(xa[i] * s0[i]);
        h[i] = hv;
        const float f = (float)hv;
        acc += f * f;
    }
#pragma unroll
    for (int i = 0; i < 4; ++i) {
        const _Float16 hv = (_Float16)(xb[i] * s1[i]);
        h[4 + i] = hv;
        const float f = (float)hv;
        acc += f * f;
    }

    *(v8h*)(dst + r * D_DIM + lane * 8) = h;

#pragma unroll
    for (int off = 16; off > 0; off >>= 1)
        acc += __shfl_xor(acc, off, 32);
    if (lane == 0) nrm[r] = acc;
}

// ---------------------------------------------------------------------------
// Kernel 3: WMMA GEMM + fused RBF epilogue.
// Block = 256 threads = 8 waves (2 M-waves x 4 N-waves); block tile 128x128.
// Each wave: 4x2 tiles of 16x16, K unrolled in chunks of 32
// via v_wmma_f32_16x16x32_f16 (fp32 accumulate).
//
// Fragment layout (CDNA5 ISA 7.12.2, 16-bit A 16x32): lane L holds row
// (L&15); k-chunks at 8*(L>>4) and 16+8*(L>>4). B^T from row-major b uses the
// identical per-lane pattern.
// ---------------------------------------------------------------------------
__global__ __launch_bounds__(256) void gemm_rbf(const _Float16* __restrict__ A,
                                                const _Float16* __restrict__ B,
                                                const float* __restrict__ n1,
                                                const float* __restrict__ n2,
                                                const float* __restrict__ var_p,
                                                float* __restrict__ out) {
    const int lane = threadIdx.x & 31;
    const int w    = threadIdx.x >> 5;
    const int wm   = w >> 2;        // 0..1
    const int wn   = w & 3;         // 0..3
    const int hi   = lane >> 4;     // 0/1
    const int lo   = lane & 15;

    const long mbase = (long)blockIdx.y * 128 + wm * 64;
    const long nbase = (long)blockIdx.x * 128 + wn * 32;

    v8f acc[4][2];
#pragma unroll
    for (int mt = 0; mt < 4; ++mt)
#pragma unroll
        for (int nt = 0; nt < 2; ++nt)
            acc[mt][nt] = (v8f){0.f, 0.f, 0.f, 0.f, 0.f, 0.f, 0.f, 0.f};

    const _Float16* aptr[4];
    const _Float16* bptr[2];
#pragma unroll
    for (int mt = 0; mt < 4; ++mt)
        aptr[mt] = A + (mbase + mt * 16 + lo) * D_DIM + 8 * hi;
#pragma unroll
    for (int nt = 0; nt < 2; ++nt)
        bptr[nt] = B + (nbase + nt * 16 + lo) * D_DIM + 8 * hi;

#pragma unroll
    for (int k = 0; k < D_DIM; k += 32) {
        if (k + 32 < D_DIM) {
            __builtin_prefetch(aptr[0] + k + 32, 0, 0);   // global_prefetch_b8
            __builtin_prefetch(bptr[0] + k + 32, 0, 0);
        }
        v16h af[4], bf[2];
#pragma unroll
        for (int mt = 0; mt < 4; ++mt) {
            const v8h l0 = *(const v8h*)(aptr[mt] + k);
            const v8h l1 = *(const v8h*)(aptr[mt] + k + 16);
            af[mt] = __builtin_shufflevector(l0, l1, 0, 1, 2, 3, 4, 5, 6, 7,
                                             8, 9, 10, 11, 12, 13, 14, 15);
        }
#pragma unroll
        for (int nt = 0; nt < 2; ++nt) {
            const v8h l0 = *(const v8h*)(bptr[nt] + k);
            const v8h l1 = *(const v8h*)(bptr[nt] + k + 16);
            bf[nt] = __builtin_shufflevector(l0, l1, 0, 1, 2, 3, 4, 5, 6, 7,
                                             8, 9, 10, 11, 12, 13, 14, 15);
        }
#pragma unroll
        for (int mt = 0; mt < 4; ++mt)
#pragma unroll
            for (int nt = 0; nt < 2; ++nt)
                acc[mt][nt] = __builtin_amdgcn_wmma_f32_16x16x32_f16(
                    /*neg_a=*/false, af[mt], /*neg_b=*/false, bf[nt],
                    /*c_mod=*/(short)0, acc[mt][nt],
                    /*reuse_a=*/false, /*reuse_b=*/false);
    }

    const float variance = var_p[0];

    // Epilogue: C/D layout -> element i of v8f is row (i + 8*hi), col (lane&15).
#pragma unroll
    for (int mt = 0; mt < 4; ++mt) {
        float rn[8];
#pragma unroll
        for (int i = 0; i < 8; ++i)
            rn[i] = n1[mbase + mt * 16 + 8 * hi + i];
#pragma unroll
        for (int nt = 0; nt < 2; ++nt) {
            const long n  = nbase + nt * 16 + lo;
            const float cn = n2[n];
#pragma unroll
            for (int i = 0; i < 8; ++i) {
                const long m = mbase + mt * 16 + 8 * hi + i;
                float sq = rn[i] + cn - 2.0f * acc[mt][nt][i];
                sq = fmaxf(sq, 0.0f);
                out[m * (long)M_ROWS + n] = variance * __expf(-0.5f * sq);
            }
        }
    }
}

// ---------------------------------------------------------------------------
// Host-side launcher (graph-capture safe: only kernel launches on `stream`).
// ---------------------------------------------------------------------------
extern "C" void kernel_launch(void* const* d_in, const int* in_sizes, int n_in,
                              void* d_out, int out_size, void* d_ws, size_t ws_size,
                              hipStream_t stream) {
    (void)in_sizes; (void)n_in; (void)out_size; (void)ws_size;

    const float* x1           = (const float*)d_in[0];   // (8192, 256)
    const float* x2           = (const float*)d_in[1];   // (8192, 256)
    const float* alpha_raw    = (const float*)d_in[2];   // (256,)
    const float* variance_raw = (const float*)d_in[3];   // (1,)
    float* out = (float*)d_out;                          // (8192, 8192)

    char* ws = (char*)d_ws;
    float*     sa  = (float*)(ws);                                   // 1 KB
    float*     var = (float*)(ws + 1024);                            // 4 B
    float*     n1  = (float*)(ws + 4096);                            // 32 KB
    float*     n2  = (float*)(ws + 4096 + 32768);                    // 32 KB
    _Float16*  a_h = (_Float16*)(ws + 131072);                       // 4 MB
    _Float16*  b_h = (_Float16*)(ws + 131072 +
                                 (size_t)N_ROWS * D_DIM * sizeof(_Float16)); // 4 MB

    prep_alpha<<<1, 256, 0, stream>>>(alpha_raw, variance_raw, sa, var);

    // 16384 rows, 8 rows (waves) per block
    scale_convert<<<(N_ROWS + M_ROWS) / 8, 256, 0, stream>>>(x1, x2, sa,
                                                             a_h, b_h, n1, n2);

    dim3 grid(M_ROWS / 128, N_ROWS / 128);   // 64 x 64 blocks of 128x128
    gemm_rbf<<<grid, 256, 0, stream>>>(a_h, b_h, n1, n2, var, out);
}
